// MXSelfAttention_45775761441069
// MI455X (gfx1250) — matrix operand order repaired and checked by reading, not compiled, not run on hardware
//
#include <hip/hip_runtime.h>
#include <hip/hip_bf16.h>

typedef __attribute__((ext_vector_type(16))) __bf16 v16bf;
typedef __attribute__((ext_vector_type(8)))  float  v8f;

#define NHEADS 16
#define SEQ    2048
#define DIM    1024
#define HD     64
#define TOPK   32

// ---------- helpers ----------

__device__ __forceinline__ unsigned short f2bf_rne(float f) {
    unsigned int x = __float_as_uint(f);
    unsigned int r = x + 0x7fffu + ((x >> 16) & 1u);
    return (unsigned short)(r >> 16);
}

// A-matrix (16x32 bf16 tile, MxK): lane l holds row m0+(l&15).
// lanes 0-15: elems 0..7 -> K kk+0..7,  elems 8..15 -> K kk+16..23
// lanes 16-31: elems 0..7 -> K kk+8..15, elems 8..15 -> K kk+24..31
__device__ __forceinline__ v16bf load_A16x32(const unsigned short* __restrict__ base,
                                             int rstride, int m0, int kk, int lane) {
    int m  = m0 + (lane & 15);
    int kb = kk + ((lane < 16) ? 0 : 8);
    const unsigned short* p = base + (size_t)m * rstride + kb;
    union { v16bf v; uint4 q[2]; } u;
    u.q[0] = *(const uint4*)(p);
    u.q[1] = *(const uint4*)(p + 16);
    return u.v;
}

// B-matrix (32x16 bf16 tile, KxN, sourced from row-major Mat[n][k]): lane l holds col n0+(l&15).
// lanes 0-15: elems 0..7 -> K kk+0..7, elems 8..15 -> K kk+16..23 ; lanes 16-31: elems 0..7 -> K kk+8..15, elems 8..15 -> K kk+24..31
__device__ __forceinline__ v16bf load_B32x16(const unsigned short* __restrict__ base,
                                             int rstride, int n0, int kk, int lane) {
    int n  = n0 + (lane & 15);
    int kb = kk + ((lane < 16) ? 0 : 8);
    const unsigned short* p = base + (size_t)n * rstride + kb;
    union { v16bf v; uint4 q[2]; } u;
    u.q[0] = *(const uint4*)(p);
    u.q[1] = *(const uint4*)(p + 16);
    return u.v;
}

#define WMMA_BF16(A, B, C) \
    __builtin_amdgcn_wmma_f32_16x16x32_bf16(false, (A), false, (B), (short)0, (C), false, false)

// ---------- kernels ----------

__global__ void f32_to_bf16_kernel(const float* __restrict__ in,
                                   unsigned short* __restrict__ out, int n) {
    int i = blockIdx.x * 256 + threadIdx.x;
    if (i < n) out[i] = f2bf_rne(in[i]);
}

// Out[m, o] = sum_k Xb[m,k] * Wb[o,k] + bias[o]   (torch Linear: x @ W.T + b)
// One wave computes a 32x32 output super-tile (2x2 register blocking: each
// operand fragment feeds 2 WMMAs -> 2 b128 loads per matrix op).
// mode 0: Q -> bf16 [B,H,N,D]          mode 1: K -> bf16 [B,H,N,D] scaled by 1/8
// mode 2: V -> f32  [B,H,N,D]          mode 3: O -> f32  [M, 1024]
__global__ void gemm_proj_kernel(const unsigned short* __restrict__ Xb,
                                 const unsigned short* __restrict__ Wb,
                                 const float* __restrict__ bias,
                                 unsigned short* __restrict__ out_bf,
                                 float* __restrict__ out_f,
                                 int mode) {
    const int wave = threadIdx.x >> 5;
    const int lane = threadIdx.x & 31;
    const int tile = blockIdx.x * 8 + wave;     // 32x32 super-tiles
    const int o0   = (tile & 31) * 32;          // DIM/32 = 32 o super-tiles
    const int m0   = (tile >> 5) * 32;          // M/32  = 128 m super-tiles

    v8f acc00 = {}, acc01 = {}, acc10 = {}, acc11 = {};
    for (int kk = 0; kk < DIM; kk += 32) {
        v16bf a0 = load_A16x32(Xb, DIM, m0,      kk, lane);
        v16bf a1 = load_A16x32(Xb, DIM, m0 + 16, kk, lane);
        v16bf b0 = load_B32x16(Wb, DIM, o0,      kk, lane);
        v16bf b1 = load_B32x16(Wb, DIM, o0 + 16, kk, lane);
        acc00 = WMMA_BF16(a0, b0, acc00);
        acc01 = WMMA_BF16(a0, b1, acc01);
        acc10 = WMMA_BF16(a1, b0, acc10);
        acc11 = WMMA_BF16(a1, b1, acc11);
    }

    const int moff = (lane < 16) ? 0 : 8;
    const int ncol = lane & 15;
    for (int half = 0; half < 4; ++half) {      // {acc00,acc01,acc10,acc11}
        v8f acc = (half == 0) ? acc00 : (half == 1) ? acc01 : (half == 2) ? acc10 : acc11;
        int mb = m0 + ((half >> 1) ? 16 : 0);
        int n  = o0 + ((half & 1) ? 16 : 0) + ncol;
        float bv = bias[n];
        for (int r = 0; r < 8; ++r) {
            int m = mb + r + moff;
            float val = acc[r] + bv;
            if (mode == 3) {
                out_f[(size_t)m * DIM + n] = val;
            } else {
                int h = n >> 6, d = n & 63;
                int bb = m >> 11, tok = m & (SEQ - 1);
                size_t idx = (((size_t)(bb * NHEADS + h)) * SEQ + tok) * HD + d;
                if (mode == 0)      out_bf[idx] = f2bf_rne(val);
                else if (mode == 1) out_bf[idx] = f2bf_rne(val * 0.125f);  // 1/sqrt(64)
                else                out_f[idx]  = val;
            }
        }
    }
}

// One block per (b,h, 16-row tile). 512 threads = 16 waves.
// Phase 1: WMMA scores 16x2048 into LDS (128 KB).
// Phase 2: register-resident iterative top-32 + softmax (wave w owns row w).
// Phase 3: sparse PV gather (coalesced).
__global__ void attn_topk_kernel(const unsigned short* __restrict__ Qb,
                                 const unsigned short* __restrict__ Kb,
                                 const float* __restrict__ Vf,
                                 float* __restrict__ X2) {
    extern __shared__ float sc[];                      // 16 * 2048 floats
    const int bh   = blockIdx.x >> 7;                  // 128 row-tiles per (b,h)
    const int row0 = (blockIdx.x & 127) * 16;
    const int wave = threadIdx.x >> 5;
    const int lane = threadIdx.x & 31;

    const unsigned short* Qbase = Qb + (size_t)bh * SEQ * HD;
    const unsigned short* Kbase = Kb + (size_t)bh * SEQ * HD;
    const float*          Vbase = Vf + (size_t)bh * SEQ * HD;

    // ---- Phase 1: scores via WMMA ----
    v16bf a0 = load_A16x32(Qbase, HD, row0, 0,  lane);
    v16bf a1 = load_A16x32(Qbase, HD, row0, 32, lane);
    for (int ct = wave; ct < SEQ / 16; ct += 16) {
        int c0 = ct * 16;
        if (ct + 16 < SEQ / 16)                         // stream next K tile into cache
            __builtin_prefetch(Kbase + (size_t)(c0 + 256) * HD, 0, 1);
        v16bf b0 = load_B32x16(Kbase, HD, c0, 0,  lane);
        v16bf b1 = load_B32x16(Kbase, HD, c0, 32, lane);
        v8f s = {};
        s = WMMA_BF16(a0, b0, s);
        s = WMMA_BF16(a1, b1, s);
        int nn   = lane & 15;
        int moff = (lane < 16) ? 0 : 8;
        for (int r = 0; r < 8; ++r)
            sc[(r + moff) * SEQ + c0 + nn] = s[r];
    }
    __syncthreads();

    // ---- Phase 2: top-32 (lane owns cols j with j%32==lane; slice kept in VGPRs) ----
    const float* srow = sc + wave * SEQ;
    float v[SEQ / 32];
#pragma unroll
    for (int t = 0; t < SEQ / 32; ++t) v[t] = srow[t * 32 + lane];

    unsigned long long taken = 0ull;
    float selv = -3.4e38f;
    int   seli = 0;
    for (int round = 0; round < TOPK; ++round) {
        float best = -3.4e38f;
        int   bt   = SEQ / 32;
#pragma unroll
        for (int t = 0; t < SEQ / 32; ++t) {           // pure-VALU masked argmax
            bool live = ((taken >> t) & 1ull) == 0ull;
            if (live && v[t] > best) { best = v[t]; bt = t; }
        }
        int bi = bt * 32 + lane;                       // global column index
        for (int off = 16; off > 0; off >>= 1) {       // argmax reduce, low-index tiebreak
            float ov = __shfl_xor(best, off, 32);
            int   oi = __shfl_xor(bi,   off, 32);
            if (ov > best || (ov == best && oi < bi)) { best = ov; bi = oi; }
        }
        if (lane == round) { selv = best; seli = bi; }
        if ((bi & 31) == lane) taken |= (1ull << (bi >> 5));
    }

    float mx = __shfl(selv, 0, 32);
    float e  = __expf(selv - mx);
    float ssum = e;
    for (int off = 16; off > 0; off >>= 1) ssum += __shfl_xor(ssum, off, 32);
    float p = e / ssum;

    // ---- Phase 3: out = sum_j p_j * V[idx_j, :]  (coalesced gather) ----
    float o0 = 0.f, o1 = 0.f;
    for (int j = 0; j < TOPK; ++j) {
        float pj = __shfl(p,    j, 32);
        int   ij = __shfl(seli, j, 32);
        const float* vr = Vbase + (size_t)ij * HD;
        o0 += pj * vr[lane];
        o1 += pj * vr[lane + 32];
    }
    int b = bh >> 4, h = bh & 15;
    float* orow = X2 + (((size_t)b * SEQ) + (row0 + wave)) * DIM + h * HD;
    orow[lane]      = o0;
    orow[lane + 32] = o1;
}

// ---------- host launch ----------

extern "C" void kernel_launch(void* const* d_in, const int* in_sizes, int n_in,
                              void* d_out, int out_size, void* d_ws, size_t ws_size,
                              hipStream_t stream) {
    const float* X  = (const float*)d_in[0];
    const float* Wq = (const float*)d_in[1];
    const float* bq = (const float*)d_in[2];
    const float* Wk = (const float*)d_in[3];
    const float* bk = (const float*)d_in[4];
    const float* Wv = (const float*)d_in[5];
    const float* bv = (const float*)d_in[6];
    const float* Wo = (const float*)d_in[7];
    const float* bo = (const float*)d_in[8];
    float* out = (float*)d_out;

    const int M = 2 * SEQ;                    // 4096 tokens
    const size_t nX = (size_t)M * DIM;        // 4,194,304
    const size_t nW = (size_t)DIM * DIM;      // 1,048,576

    char* ws = (char*)d_ws;
    size_t off = 0;
    unsigned short* Xb  = (unsigned short*)(ws + off); off += nX * 2;
    unsigned short* Wqb = (unsigned short*)(ws + off); off += nW * 2;
    unsigned short* Wkb = (unsigned short*)(ws + off); off += nW * 2;
    unsigned short* Wvb = (unsigned short*)(ws + off); off += nW * 2;
    unsigned short* Wob = (unsigned short*)(ws + off); off += nW * 2;
    unsigned short* Qb  = (unsigned short*)(ws + off); off += nX * 2;
    unsigned short* Kb  = (unsigned short*)(ws + off); off += nX * 2;
    float*          Vf  = (float*)(ws + off);          off += nX * 4;
    float*          X2  = (float*)(ws + off);          off += nX * 4;
    unsigned short* X2b = (unsigned short*)(ws + off); off += nX * 2;

    // 1. bf16 conversions
    f32_to_bf16_kernel<<<(nX + 255) / 256, 256, 0, stream>>>(X,  Xb,  (int)nX);
    f32_to_bf16_kernel<<<(nW + 255) / 256, 256, 0, stream>>>(Wq, Wqb, (int)nW);
    f32_to_bf16_kernel<<<(nW + 255) / 256, 256, 0, stream>>>(Wk, Wkb, (int)nW);
    f32_to_bf16_kernel<<<(nW + 255) / 256, 256, 0, stream>>>(Wv, Wvb, (int)nW);
    f32_to_bf16_kernel<<<(nW + 255) / 256, 256, 0, stream>>>(Wo, Wob, (int)nW);

    // 2. QKV projections (WMMA). (M/32)*(DIM/32) = 4096 super-tiles, 8 per block
    dim3 gproj(4096 / 8);
    gemm_proj_kernel<<<gproj, 256, 0, stream>>>(Xb, Wqb, bq, Qb, nullptr, 0);
    gemm_proj_kernel<<<gproj, 256, 0, stream>>>(Xb, Wkb, bk, Kb, nullptr, 1);
    gemm_proj_kernel<<<gproj, 256, 0, stream>>>(Xb, Wvb, bv, nullptr, Vf, 2);

    // 3. fused scores + top-k + softmax + PV : B*H*(N/16) = 4096 blocks, 128 KB LDS
    attn_topk_kernel<<<2 * NHEADS * (SEQ / 16), 512, 16 * SEQ * sizeof(float), stream>>>(
        Qb, Kb, Vf, X2);

    // 4. output projection
    f32_to_bf16_kernel<<<(nX + 255) / 256, 256, 0, stream>>>(X2, X2b, (int)nX);
    gemm_proj_kernel<<<gproj, 256, 0, stream>>>(X2b, Wob, bo, nullptr, out, 3);
}
